// DEQFRR_34282428957310
// MI455X (gfx1250) — matrix operand order repaired and checked by reading, not compiled, not run on hardware
//
#include <hip/hip_runtime.h>
#include <cstddef>

// =====================================================================
// MI455X (gfx1250) implementation of the DEQ/fractal transformer forward.
//  - bf16 WMMA (v_wmma_f32_16x16x32_bf16) for every GEMM + attention
//  - GEMM: double-buffered LDS staging via GLOBAL_LOAD_ASYNC_TO_LDS_B128
//    (ASYNCcnt pipeline, s_wait_asynccnt), WMMA fed from LDS
//  - flash-style attention (no [B,NH,S,S] score materialization)
//  - weights converted to bf16 once per call; whole working set is
//    L2-resident (192MB L2), so the solve runs out of L2 after first touch
//  - device-side fixed-point convergence flag; all 12 iterations are in
//    the graph, kernels early-exit once converged (deterministic)
// =====================================================================

typedef __attribute__((ext_vector_type(16))) __bf16        v16bf;
typedef __attribute__((ext_vector_type(8)))  float          v8f;
typedef __attribute__((ext_vector_type(8)))  unsigned short u16x8;
typedef __attribute__((ext_vector_type(16))) unsigned short u16x16;

#define H_DIM   1024
#define M_ROWS  2048
#define S_LEN   1024
#define NHEADS  16
#define DHEAD   64
#define VOCAB   32000
#define N_ITER  12

// ---------------------------- helpers --------------------------------
__device__ __forceinline__ unsigned short f2bf(float f){
  unsigned int u = __float_as_uint(f);
  u += 0x7fffu + ((u >> 16) & 1u);           // round-to-nearest-even
  return (unsigned short)(u >> 16);
}
__device__ __forceinline__ v8f vzero8(){
  v8f z;
#pragma unroll
  for (int i = 0; i < 8; i++) z[i] = 0.f;
  return z;
}
__device__ __forceinline__ v8f wmma_bf16(v16bf a, v16bf b, v8f c){
  // (neg_a, A, neg_b, B, c_mod, C, reuse_a, reuse_b)
  return __builtin_amdgcn_wmma_f32_16x16x32_bf16(false, a, false, b, (short)0, c, false, false);
}
// A-fragment (16x32, M x K): lane m (0-15) + half (lane>>4):
//   elems 0-7  = row m, K = k0 + half*16 .. +7
//   elems 8-15 = row m, K = k0 + half*16 + 8 .. +15
__device__ __forceinline__ v16bf load_a_frag(const unsigned short* rowbase, int ld, int k0, int lane){
  const int m = lane & 15, half = lane >> 4;
  const unsigned short* p = rowbase + (size_t)m * ld + k0;
  u16x8 lo = *(const u16x8*)(p + half * 16);
  u16x8 hi = *(const u16x8*)(p + half * 16 + 8);
  u16x16 c;
#pragma unroll
  for (int i = 0; i < 8; i++){ c[i] = lo[i]; c[i + 8] = hi[i]; }
  return __builtin_bit_cast(v16bf, c);
}
// B-fragment (32x16, K x N) for C = A @ W^T: lane n holds row n of W,
// 16 consecutive K values starting at k0 + half*16 (single 32B load).
__device__ __forceinline__ v16bf load_b_frag(const unsigned short* colbase, int ld, int k0, int lane){
  const int n = lane & 15, half = lane >> 4;
  u16x16 c = *(const u16x16*)(colbase + (size_t)n * ld + k0 + half * 16);
  return __builtin_bit_cast(v16bf, c);
}
__device__ __forceinline__ float gelu_tanh(float x){
  float x3 = x * x * x;
  return 0.5f * x * (1.f + tanhf(0.7978845608028654f * (x + 0.044715f * x3)));
}
__device__ __forceinline__ float block_sum_256(float v, float* red){
#pragma unroll
  for (int m = 16; m > 0; m >>= 1) v += __shfl_xor(v, m);
  __syncthreads();
  if ((threadIdx.x & 31) == 0) red[threadIdx.x >> 5] = v;
  __syncthreads();
  return red[0] + red[1] + red[2] + red[3] + red[4] + red[5] + red[6] + red[7];
}

// ---- CDNA5 async copy: global -> LDS, tracked by ASYNCcnt ------------
// Copies a 128-row x 32-col bf16 tile (8KB) with 256 threads x 2 x 16B.
// vdst operand = LDS byte address VGPR (generic LDS ptr truncates to the
// addr[31:0] LDS offset per the aperture rules), vaddr = 64b global addr.
__device__ __forceinline__ void async_tile_load(const unsigned short* gbase, int ld,
                                                int row0, int k0,
                                                unsigned lds_base, int tid)
{
#pragma unroll
  for (int j = 0; j < 2; j++){
    const int c   = tid + j * 256;            // 0..511 16B segments
    const int row = c >> 2;
    const int seg = c & 3;
    const void* gp = (const void*)(gbase + (size_t)(row0 + row) * ld + k0 + seg * 8);
    const unsigned lo = lds_base + (unsigned)(row * 64 + seg * 16);
    asm volatile("global_load_async_to_lds_b128 %0, %1, off"
                 :: "v"(lo), "v"(gp) : "memory");
  }
}
__device__ __forceinline__ void wait_async_le4(){
  asm volatile("s_wait_asynccnt 0x4" ::: "memory");
}
__device__ __forceinline__ void wait_async_0(){
  asm volatile("s_wait_asynccnt 0x0" ::: "memory");
}

// ---------------------- elementwise / setup kernels -------------------
__global__ __launch_bounds__(256) void cvt_kernel(const float* __restrict__ src,
                                                  unsigned short* __restrict__ dst, int n){
  int base = blockIdx.x * 1024 + threadIdx.x;
#pragma unroll
  for (int j = 0; j < 4; j++){
    int idx = base + j * 256;
    if (idx < n) dst[idx] = f2bf(src[idx]);
  }
}

__global__ __launch_bounds__(256) void gather_kernel(const int* __restrict__ tokens,
                                                     const float* __restrict__ embed,
                                                     unsigned short* __restrict__ out){
  const int m = blockIdx.x;                       // 0..2047  (b*1024 + s)
  const size_t erow = (size_t)tokens[m] * H_DIM;
  const size_t orow = (size_t)m * H_DIM;
#pragma unroll
  for (int j = 0; j < 4; j++){
    int c = threadIdx.x + j * 256;
    out[orow + c] = f2bf(embed[erow + c]);
  }
}

__global__ __launch_bounds__(256) void init_kernel(float* __restrict__ Z, float* __restrict__ scal){
  const size_t base = (size_t)blockIdx.x * 1024 + threadIdx.x;
#pragma unroll
  for (int j = 0; j < 4; j++) Z[base + j * 256] = 0.f;
  if (blockIdx.x == 0 && threadIdx.x == 0){
    scal[0] = 0.f; scal[1] = 0.f;
    ((int*)scal)[2] = 0;        // converged flag
    ((int*)scal)[3] = N_ITER;   // n_iters default
  }
}

// ------------------------------ GEMM ---------------------------------
// C[M,N] = A[M,K] @ W[N,K]^T, bf16 in, f32 accumulate, WMMA 16x16x32.
// 256 threads = 8 waves; block tile 128x128; wave tile 64x32 (4x2 WMMA
// tiles). K-tiles of 32 are staged into a double-buffered LDS tile with
// GLOBAL_LOAD_ASYNC_TO_LDS_B128; copy of tile k+1 overlaps WMMA of tile k.
// EPI: 0 f32 store | 1 bf16 store | 2 bf16 transposed store (ld=ldt)
//      3 f32 store of acc+Y        | 4 bf16 store of gelu(acc)
template<int EPI>
__global__ __launch_bounds__(256) void gemm_kernel(
    const unsigned short* __restrict__ A, const unsigned short* __restrict__ W,
    const float* __restrict__ Y, float* __restrict__ Cf, unsigned short* __restrict__ Cb,
    int Mm, int Nn, int Kk, int ldt, const int* flag)
{
  if (flag && *flag) return;
  __shared__ __align__(64) unsigned short As[2][128 * 32];
  __shared__ __align__(64) unsigned short Bs[2][128 * 32];
  const int tid  = threadIdx.x;
  const int lane = tid & 31, wave = tid >> 5;
  const int wm = wave & 1, wn = wave >> 1;
  const int rowB0 = blockIdx.y * 128;
  const int colB0 = blockIdx.x * 128;
  (void)Mm;

  const unsigned asOff[2] = { (unsigned)(size_t)&As[0][0], (unsigned)(size_t)&As[1][0] };
  const unsigned bsOff[2] = { (unsigned)(size_t)&Bs[0][0], (unsigned)(size_t)&Bs[1][0] };

  v8f acc[4][2];
#pragma unroll
  for (int i = 0; i < 4; i++)
#pragma unroll
    for (int j = 0; j < 2; j++) acc[i][j] = vzero8();

  // prologue: stage k-tile 0 (4 async b128 per thread per stage)
  async_tile_load(A, Kk, rowB0, 0, asOff[0], tid);
  async_tile_load(W, Kk, colB0, 0, bsOff[0], tid);

  const int KT = Kk / 32;
  for (int kt = 0; kt < KT; kt++){
    const int cur = kt & 1;
    if (kt + 1 < KT){
      async_tile_load(A, Kk, rowB0, (kt + 1) * 32, asOff[cur ^ 1], tid);
      async_tile_load(W, Kk, colB0, (kt + 1) * 32, bsOff[cur ^ 1], tid);
      wait_async_le4();              // oldest 4 (= stage kt) retired, in order
    } else {
      wait_async_0();
    }
    __syncthreads();                  // cross-wave visibility of stage kt

    v16bf af[4], bfrag[2];
#pragma unroll
    for (int tm = 0; tm < 4; tm++)
      af[tm] = load_a_frag(&As[cur][(wm * 64 + tm * 16) * 32], 32, 0, lane);
#pragma unroll
    for (int tn = 0; tn < 2; tn++)
      bfrag[tn] = load_b_frag(&Bs[cur][(wn * 32 + tn * 16) * 32], 32, 0, lane);
#pragma unroll
    for (int tm = 0; tm < 4; tm++)
#pragma unroll
      for (int tn = 0; tn < 2; tn++)
        acc[tm][tn] = wmma_bf16(af[tm], bfrag[tn], acc[tm][tn]);
    __syncthreads();                  // all reads done before kt+2 overwrites
  }

  // C layout: VGPR r, lanes 0-15 -> row r, lanes 16-31 -> row r+8; col = lane&15
#pragma unroll
  for (int tm = 0; tm < 4; tm++){
#pragma unroll
    for (int tn = 0; tn < 2; tn++){
      const int col = colB0 + wn * 32 + tn * 16 + (lane & 15);
      const int rb  = rowB0 + wm * 64 + tm * 16 + (lane >> 4) * 8;
#pragma unroll
      for (int r = 0; r < 8; r++){
        const size_t idx = (size_t)(rb + r) * Nn + col;
        float v = acc[tm][tn][r];
        if (EPI == 0)      Cf[idx] = v;
        else if (EPI == 1) Cb[idx] = f2bf(v);
        else if (EPI == 2) Cb[(size_t)col * ldt + (rb + r)] = f2bf(v);
        else if (EPI == 3) Cf[idx] = v + Y[idx];
        else               Cb[idx] = f2bf(gelu_tanh(v));
      }
    }
  }
}

// --------------------------- flash attention --------------------------
// grid (S/64, NH, B), 128 threads = 4 waves; each wave owns 16 query rows.
// Q,K: [M,H] bf16 row-major (head = column slice); Vt: [H, M] bf16 (V^T).
// Online softmax in C-layout halves via shfl_xor; P relayout via wave-LDS.
__global__ __launch_bounds__(128) void flash_kernel(
    const unsigned short* __restrict__ Q, const unsigned short* __restrict__ Km,
    const unsigned short* __restrict__ Vt, unsigned short* __restrict__ O,
    const int* flag)
{
  if (flag && *flag) return;
  const int lane = threadIdx.x & 31, wave = threadIdx.x >> 5;
  const int lm = lane & 15, half = lane >> 4;
  const int qb0  = blockIdx.x * 64 + wave * 16;   // query tile base (within seq)
  const int hb   = blockIdx.y * DHEAD;            // head column base
  const size_t rowb = (size_t)blockIdx.z * S_LEN; // batch row offset

  __shared__ __align__(16) unsigned short psh[4][16][32];

  // Q fragments over dh=64 (two K-chunks of 32)
  v16bf qf[2];
#pragma unroll
  for (int kk = 0; kk < 2; kk++){
    const unsigned short* qp = Q + (rowb + qb0 + lm) * H_DIM + hb + kk * 32;
    u16x8 lo = *(const u16x8*)(qp + half * 16);
    u16x8 hi = *(const u16x8*)(qp + half * 16 + 8);
    u16x16 c;
#pragma unroll
    for (int i = 0; i < 8; i++){ c[i] = lo[i]; c[i + 8] = hi[i]; }
    qf[kk] = __builtin_bit_cast(v16bf, c);
  }

  v8f oacc[4];
#pragma unroll
  for (int dt = 0; dt < 4; dt++) oacc[dt] = vzero8();
  float m8[8], l8[8];
#pragma unroll
  for (int r = 0; r < 8; r++){ m8[r] = -1e30f; l8[r] = 0.f; }

  const int kend = blockIdx.x * 64 + 64;          // causal bound for this block
  for (int j0 = 0; j0 < kend; j0 += 32){
    // ---- scores S[16q x 32k] = Q @ K^T / sqrt(dh) ----
    v8f sacc[2];
#pragma unroll
    for (int tn = 0; tn < 2; tn++) sacc[tn] = vzero8();
#pragma unroll
    for (int tn = 0; tn < 2; tn++){
#pragma unroll
      for (int kk = 0; kk < 2; kk++){
        const unsigned short* kp =
            Km + (rowb + j0 + tn * 16 + lm) * H_DIM + hb + kk * 32 + half * 16;
        u16x16 kc = *(const u16x16*)kp;
        sacc[tn] = wmma_bf16(qf[kk], __builtin_bit_cast(v16bf, kc), sacc[tn]);
      }
    }
    // ---- scale + causal mask ----
#pragma unroll
    for (int tn = 0; tn < 2; tn++){
      const int key = j0 + tn * 16 + lm;
#pragma unroll
      for (int r = 0; r < 8; r++){
        const int qr = qb0 + r + 8 * half;
        float s = sacc[tn][r] * 0.125f;             // 1/sqrt(64)
        sacc[tn][r] = (key <= qr) ? s : -1e30f;
      }
    }
    // ---- online softmax stats (reduce over 16 lanes of each half) ----
    float mnew[8], scl[8];
#pragma unroll
    for (int r = 0; r < 8; r++){
      float v = fmaxf(sacc[0][r], sacc[1][r]);
      v = fmaxf(v, __shfl_xor(v, 1)); v = fmaxf(v, __shfl_xor(v, 2));
      v = fmaxf(v, __shfl_xor(v, 4)); v = fmaxf(v, __shfl_xor(v, 8));
      mnew[r] = fmaxf(m8[r], v);
      scl[r]  = __expf(m8[r] - mnew[r]);
      m8[r]   = mnew[r];
    }
#pragma unroll
    for (int tn = 0; tn < 2; tn++)
#pragma unroll
      for (int r = 0; r < 8; r++)
        sacc[tn][r] = __expf(sacc[tn][r] - mnew[r]);
#pragma unroll
    for (int r = 0; r < 8; r++){
      float sum = sacc[0][r] + sacc[1][r];
      sum += __shfl_xor(sum, 1); sum += __shfl_xor(sum, 2);
      sum += __shfl_xor(sum, 4); sum += __shfl_xor(sum, 8);
      l8[r] = l8[r] * scl[r] + sum;
    }
#pragma unroll
    for (int dt = 0; dt < 4; dt++)
#pragma unroll
      for (int r = 0; r < 8; r++) oacc[dt][r] *= scl[r];

    // ---- P: C-layout -> A-layout via wave-private LDS tile ----
#pragma unroll
    for (int tn = 0; tn < 2; tn++)
#pragma unroll
      for (int r = 0; r < 8; r++)
        psh[wave][r + 8 * half][tn * 16 + lm] = f2bf(sacc[tn][r]);
    __syncthreads();
    u16x16 pc;
#pragma unroll
    for (int i = 0; i < 8; i++){
      pc[i]     = psh[wave][lm][half * 16 + i];
      pc[i + 8] = psh[wave][lm][half * 16 + 8 + i];
    }
    v16bf pf = __builtin_bit_cast(v16bf, pc);
    // ---- O += P @ V  (V consumed from pre-transposed Vt, contiguous) ----
#pragma unroll
    for (int dt = 0; dt < 4; dt++){
      const unsigned short* vp =
          Vt + (size_t)(hb + dt * 16 + lm) * M_ROWS + rowb + j0 + half * 16;
      u16x16 vc = *(const u16x16*)vp;
      oacc[dt] = wmma_bf16(pf, __builtin_bit_cast(v16bf, vc), oacc[dt]);
    }
    __syncthreads();
  }
  // ---- normalize + store ----
#pragma unroll
  for (int dt = 0; dt < 4; dt++){
    const int col = hb + dt * 16 + lm;
#pragma unroll
    for (int r = 0; r < 8; r++){
      const int qr = qb0 + r + 8 * half;
      O[(rowb + qr) * H_DIM + col] = f2bf(oacc[dt][r] / l8[r]);
    }
  }
}

// ------------------------- norm / update kernels ----------------------
__global__ __launch_bounds__(256) void film_norm_kernel(
    const float* __restrict__ X, const float* __restrict__ Add,
    float* __restrict__ Ysave, const float* __restrict__ Gw,
    const float* __restrict__ Bw, unsigned short* __restrict__ Out,
    const int* flag)
{
  if (flag && *flag) return;
  __shared__ float red[8];
  const size_t base = (size_t)blockIdx.x * H_DIM;
  float v[4]; float ss = 0.f;
#pragma unroll
  for (int i = 0; i < 4; i++){
    int c = threadIdx.x + i * 256;
    float x = X[base + c];
    if (Add) x += Add[base + c];
    v[i] = x; ss += x * x;
  }
  float tot = block_sum_256(ss, red);
  float inv = rsqrtf(tot * (1.f / H_DIM) + 1e-6f);
#pragma unroll
  for (int i = 0; i < 4; i++){
    int c = threadIdx.x + i * 256;
    float o = v[i] * inv * Gw[c] + (Bw ? Bw[c] : 0.f);
    Out[base + c] = f2bf(o);
    if (Ysave) Ysave[base + c] = v[i];
  }
}

__global__ __launch_bounds__(256) void update_kernel(
    float* __restrict__ Z, const float* __restrict__ Onew,
    const float* __restrict__ step_scale, int k, float* scal, const int* flag)
{
  if (flag && *flag) return;
  __shared__ float red[8];
  const float step = tanhf(step_scale[k < 39 ? k : 39]);
  const size_t base = (size_t)blockIdx.x * H_DIM;
  float dsq = 0.f, zsq = 0.f;
#pragma unroll
  for (int i = 0; i < 4; i++){
    size_t idx = base + threadIdx.x + i * 256;
    float z = Z[idx];
    float d = step * (Onew[idx] - z);
    Z[idx] = z + d;                       // in-place: z_new = z + step*delta
    dsq += d * d; zsq += z * z;
  }
  float dtot = block_sum_256(dsq, red);
  float ztot = block_sum_256(zsq, red);
  if (threadIdx.x == 0){
    atomicAdd(&scal[0], sqrtf(dtot));
    atomicAdd(&scal[1], sqrtf(ztot));
  }
}

__global__ void finalize_kernel(float* scal, int k){
  if (threadIdx.x == 0 && blockIdx.x == 0){
    int* ip = (int*)scal;
    if (ip[2] == 0){
      float dn = scal[0] * (1.f / M_ROWS);
      float zn = scal[1] * (1.f / M_ROWS);
      float res = dn / (1e-5f + zn);
      ip[3] = k + 1;                       // n_iters
      if (res < 1e-3f) ip[2] = 1;          // converged
    }
    scal[0] = 0.f; scal[1] = 0.f;
  }
}

__global__ __launch_bounds__(256) void final_update_kernel(
    const float* __restrict__ Z, const float* __restrict__ Onew,
    const float* __restrict__ step_scale, const float* __restrict__ scal,
    unsigned short* __restrict__ Zo)
{
  const int* ip = (const int*)scal;
  int k = ip[3] - 1; k = k < 0 ? 0 : (k > 39 ? 39 : k);
  const float step = tanhf(step_scale[k]);
  const size_t base = (size_t)blockIdx.x * H_DIM;
#pragma unroll
  for (int i = 0; i < 4; i++){
    size_t idx = base + threadIdx.x + i * 256;
    Zo[idx] = f2bf(Z[idx] + step * (Onew[idx] - Z[idx]));
  }
}

// ============================ host driver =============================
extern "C" void kernel_launch(void* const* d_in, const int* in_sizes, int n_in,
                              void* d_out, int out_size, void* d_ws, size_t ws_size,
                              hipStream_t stream)
{
  (void)in_sizes; (void)n_in; (void)out_size; (void)ws_size;
  const int*   tokens     = (const int*)  d_in[0];
  const float* embed_w    = (const float*)d_in[1];
  const float* lm_head_w  = (const float*)d_in[2];
  const float* norm_w     = (const float*)d_in[3];
  const float* proj_in_w  = (const float*)d_in[4];
  const float* proj_out_w = (const float*)d_in[5];
  const float* gamma      = (const float*)d_in[6];
  const float* beta       = (const float*)d_in[7];
  const float* step_scale = (const float*)d_in[8];
  const float* wq = (const float*)d_in[9];
  const float* wk = (const float*)d_in[10];
  const float* wv = (const float*)d_in[11];
  const float* wo = (const float*)d_in[12];
  const float* w1 = (const float*)d_in[13];
  const float* w2 = (const float*)d_in[14];
  float* logits = (float*)d_out;

  // ---- workspace carve-up (~150MB; all bump-allocated, 256B aligned) ----
  char* wsb = (char*)d_ws;
  size_t off = 0;
  auto alloc = [&](size_t bytes) -> void* {
    void* p = wsb + off;
    off = (off + bytes + 255) & ~(size_t)255;
    return p;
  };
  const size_t HH = (size_t)H_DIM * H_DIM;
  unsigned short* wqb  = (unsigned short*)alloc(HH * 2);
  unsigned short* wkb  = (unsigned short*)alloc(HH * 2);
  unsigned short* wvb  = (unsigned short*)alloc(HH * 2);
  unsigned short* wob  = (unsigned short*)alloc(HH * 2);
  unsigned short* w1b  = (unsigned short*)alloc(HH * 2);
  unsigned short* w2b  = (unsigned short*)alloc(HH * 2);
  unsigned short* pinb = (unsigned short*)alloc(HH * 2);
  unsigned short* poutb= (unsigned short*)alloc(HH * 2);
  unsigned short* lmhb = (unsigned short*)alloc((size_t)VOCAB * H_DIM * 2);
  unsigned short* xob  = (unsigned short*)alloc((size_t)M_ROWS * H_DIM * 2);
  float* xef  = (float*)alloc((size_t)M_ROWS * H_DIM * 4);
  float* zf   = (float*)alloc((size_t)M_ROWS * H_DIM * 4);
  float* yf   = (float*)alloc((size_t)M_ROWS * H_DIM * 4);
  float* x2f  = (float*)alloc((size_t)M_ROWS * H_DIM * 4);
  float* outf = (float*)alloc((size_t)M_ROWS * H_DIM * 4);
  unsigned short* hnb  = (unsigned short*)alloc((size_t)M_ROWS * H_DIM * 2);
  unsigned short* hn2b = (unsigned short*)alloc((size_t)M_ROWS * H_DIM * 2);
  unsigned short* qb   = (unsigned short*)alloc((size_t)M_ROWS * H_DIM * 2);
  unsigned short* kb   = (unsigned short*)alloc((size_t)M_ROWS * H_DIM * 2);
  unsigned short* vtb  = (unsigned short*)alloc((size_t)M_ROWS * H_DIM * 2);
  unsigned short* aob  = (unsigned short*)alloc((size_t)M_ROWS * H_DIM * 2);
  float* scal = (float*)alloc(256);
  unsigned short* gb   = qb;    // w1 output reuses Q (dead after attention)
  unsigned short* zob  = kb;    // final z_out bf16 reuses K
  unsigned short* latb = hnb;   // final latent reuses hn
  int* flag = (int*)scal + 2;

  const dim3 blk(256), g1024(H_DIM / 128, M_ROWS / 128), gvoc(VOCAB / 128, M_ROWS / 128);

  // ---- setup: weight conversion, embedding gather, z=0, x_embed ----
  const float* wsrc[8] = {wq, wk, wv, wo, w1, w2, proj_in_w, proj_out_w};
  unsigned short* wdst[8] = {wqb, wkb, wvb, wob, w1b, w2b, pinb, poutb};
  for (int i = 0; i < 8; i++)
    cvt_kernel<<<HH / 1024, blk, 0, stream>>>(wsrc[i], wdst[i], (int)HH);
  cvt_kernel<<<((size_t)VOCAB * H_DIM) / 1024, blk, 0, stream>>>(lm_head_w, lmhb, VOCAB * H_DIM);
  gather_kernel<<<M_ROWS, blk, 0, stream>>>(tokens, embed_w, xob);
  init_kernel<<<M_ROWS, blk, 0, stream>>>(zf, scal);
  gemm_kernel<0><<<g1024, blk, 0, stream>>>(xob, pinb, nullptr, xef, nullptr,
                                            M_ROWS, H_DIM, H_DIM, 0, nullptr);

  // ---- fixed-point solve: 12 iterations with device-side early-out ----
  for (int k = 0; k < N_ITER; k++){
    film_norm_kernel<<<M_ROWS, blk, 0, stream>>>(zf, xef, yf, gamma, beta, hnb, flag);
    gemm_kernel<1><<<g1024, blk, 0, stream>>>(hnb, wqb, nullptr, nullptr, qb,
                                              M_ROWS, H_DIM, H_DIM, 0, flag);
    gemm_kernel<1><<<g1024, blk, 0, stream>>>(hnb, wkb, nullptr, nullptr, kb,
                                              M_ROWS, H_DIM, H_DIM, 0, flag);
    gemm_kernel<2><<<g1024, blk, 0, stream>>>(hnb, wvb, nullptr, nullptr, vtb,
                                              M_ROWS, H_DIM, H_DIM, M_ROWS, flag);
    flash_kernel<<<dim3(S_LEN / 64, NHEADS, 2), 128, 0, stream>>>(qb, kb, vtb, aob, flag);
    gemm_kernel<3><<<g1024, blk, 0, stream>>>(aob, wob, yf, x2f, nullptr,
                                              M_ROWS, H_DIM, H_DIM, 0, flag);
    film_norm_kernel<<<M_ROWS, blk, 0, stream>>>(x2f, nullptr, nullptr, gamma, beta, hn2b, flag);
    gemm_kernel<4><<<g1024, blk, 0, stream>>>(hn2b, w1b, nullptr, nullptr, gb,
                                              M_ROWS, H_DIM, H_DIM, 0, flag);
    gemm_kernel<3><<<g1024, blk, 0, stream>>>(gb, w2b, x2f, outf, nullptr,
                                              M_ROWS, H_DIM, H_DIM, 0, flag);
    update_kernel<<<M_ROWS, blk, 0, stream>>>(zf, outf, step_scale, k, scal, flag);
    finalize_kernel<<<1, 32, 0, stream>>>(scal, k);
  }

  // ---- phantom-gradient forward pass (always runs) ----
  film_norm_kernel<<<M_ROWS, blk, 0, stream>>>(zf, xef, yf, gamma, beta, hnb, nullptr);
  gemm_kernel<1><<<g1024, blk, 0, stream>>>(hnb, wqb, nullptr, nullptr, qb,
                                            M_ROWS, H_DIM, H_DIM, 0, nullptr);
  gemm_kernel<1><<<g1024, blk, 0, stream>>>(hnb, wkb, nullptr, nullptr, kb,
                                            M_ROWS, H_DIM, H_DIM, 0, nullptr);
  gemm_kernel<2><<<g1024, blk, 0, stream>>>(hnb, wvb, nullptr, nullptr, vtb,
                                            M_ROWS, H_DIM, H_DIM, M_ROWS, nullptr);
  flash_kernel<<<dim3(S_LEN / 64, NHEADS, 2), 128, 0, stream>>>(qb, kb, vtb, aob, nullptr);
  gemm_kernel<3><<<g1024, blk, 0, stream>>>(aob, wob, yf, x2f, nullptr,
                                            M_ROWS, H_DIM, H_DIM, 0, nullptr);
  film_norm_kernel<<<M_ROWS, blk, 0, stream>>>(x2f, nullptr, nullptr, gamma, beta, hn2b, nullptr);
  gemm_kernel<4><<<g1024, blk, 0, stream>>>(hn2b, w1b, nullptr, nullptr, gb,
                                            M_ROWS, H_DIM, H_DIM, 0, nullptr);
  gemm_kernel<3><<<g1024, blk, 0, stream>>>(gb, w2b, x2f, outf, nullptr,
                                            M_ROWS, H_DIM, H_DIM, 0, nullptr);
  final_update_kernel<<<M_ROWS, blk, 0, stream>>>(zf, outf, step_scale, scal, zob);

  // ---- head: proj_out -> RMSNorm*norm_w -> lm_head ----
  gemm_kernel<0><<<g1024, blk, 0, stream>>>(zob, poutb, nullptr, yf, nullptr,
                                            M_ROWS, H_DIM, H_DIM, 0, nullptr);
  film_norm_kernel<<<M_ROWS, blk, 0, stream>>>(yf, nullptr, nullptr, norm_w, nullptr, latb, nullptr);
  gemm_kernel<0><<<gvoc, blk, 0, stream>>>(latb, lmhb, nullptr, logits, nullptr,
                                           M_ROWS, VOCAB, H_DIM, 0, nullptr);
}
